// GENConvNet_51754355916839
// MI455X (gfx1250) — matrix operand order repaired and checked by reading, not compiled, not run on hardware
//
#include <hip/hip_runtime.h>
#include <hip/hip_bf16.h>

#define NNODES 100000
#define NEDGES 1600000
#define NGRAPHS 64
#define DIN0 128
#define DH 192
#define DOUTL 128
#define OUTDIM 10
#define GEPS 1e-7f

typedef __attribute__((ext_vector_type(16))) __bf16 v16bf;
typedef __attribute__((ext_vector_type(8)))  __bf16 v8bf;
typedef __attribute__((ext_vector_type(4)))  __bf16 v4bf;
typedef __attribute__((ext_vector_type(8)))  float  v8f;

static __device__ __forceinline__ __bf16 f2bf(float f) {
  unsigned u = __builtin_bit_cast(unsigned, f);
  unsigned r = u + 0x7FFFu + ((u >> 16) & 1u);   // round-to-nearest-even
  unsigned short h = (unsigned short)(r >> 16);
  return __builtin_bit_cast(__bf16, h);
}

static __device__ __forceinline__ void atomAddF32(float* p, float v) {
  __hip_atomic_fetch_add(p, v, __ATOMIC_RELAXED, __HIP_MEMORY_SCOPE_AGENT);
}

// ---------------- zero (float4) ----------------
__global__ void k_zero_f4(float4* __restrict__ p, long n4) {
  long i = (long)blockIdx.x * blockDim.x + threadIdx.x;
  if (i < n4) p[i] = make_float4(0.f, 0.f, 0.f, 0.f);
}

// ---------------- scatter: agg[dst] += relu(x[src]) + eps ----------------
template <int D4>
__global__ void k_scatter(const float* __restrict__ x, const int* __restrict__ src,
                          const int* __restrict__ dst, float* __restrict__ agg) {
  long t = (long)blockIdx.x * blockDim.x + threadIdx.x;
  if (t >= (long)NEDGES * D4) return;
  int e = (int)(t / D4);          // compile-time divisor -> mul/shift
  int c = (int)(t % D4);
  int s = src[e];
  int dd = dst[e];
  const float4 v = ((const float4*)(x + (long)s * (D4 * 4)))[c];
  float4 m;
  m.x = fmaxf(v.x, 0.0f) + GEPS;
  m.y = fmaxf(v.y, 0.0f) + GEPS;
  m.z = fmaxf(v.z, 0.0f) + GEPS;
  m.w = fmaxf(v.w, 0.0f) + GEPS;
  float* b = agg + (long)dd * (D4 * 4) + c * 4;
  atomAddF32(b + 0, m.x);
  atomAddF32(b + 1, m.y);
  atomAddF32(b + 2, m.z);
  atomAddF32(b + 3, m.w);
}

// ---------------- fused residual add + f32->bf16 convert (x4) ----------------
__global__ void k_addcvt(const float4* __restrict__ x, const float4* __restrict__ agg,
                         v4bf* __restrict__ h, long n4) {
  long i = (long)blockIdx.x * blockDim.x + threadIdx.x;
  if (i >= n4) return;
  float4 a = x[i];
  float4 b = agg[i];
  v4bf o;
  o[0] = f2bf(a.x + b.x);
  o[1] = f2bf(a.y + b.y);
  o[2] = f2bf(a.z + b.z);
  o[3] = f2bf(a.w + b.w);
  h[i] = o;
}

// ---------------- weight transpose + convert: Wt[dout][din] = bf16(W[din][dout]) ----
__global__ void k_wtcvt(const float* __restrict__ W, __bf16* __restrict__ Wt,
                        int din, int dout) {
  int i = blockIdx.x * blockDim.x + threadIdx.x;
  if (i >= din * dout) return;
  int r = i / dout, c = i % dout;
  Wt[(long)c * din + r] = f2bf(W[i]);
}

// ---------------- GEMM: out[N,dout] = A[N,din](bf16) @ Wt^T + bias ----------------
// One 16x32 output slab per wave (two 16x16 WMMA tiles sharing the A fragment),
// K-step 32 via v_wmma_f32_16x16x32_bf16.
__global__ void k_gemm(const __bf16* __restrict__ A, const __bf16* __restrict__ Wt,
                       const float* __restrict__ bias, float* __restrict__ out,
                       int din, int dout) {
  const int wave  = threadIdx.x >> 5;
  const int lane  = threadIdx.x & 31;
  const int waves = blockDim.x >> 5;
  const int mtile = blockIdx.x * waves + wave;
  const int mtiles = NNODES / 16;            // 6250, exact
  if (mtile >= mtiles) return;               // wave-uniform: EXEC all-ones for WMMA
  const int half  = lane >> 4;
  const int l15   = lane & 15;

  const __bf16* arow  = A + (long)(mtile * 16 + l15) * din;
  const int col0      = blockIdx.y * 32 + l15;
  const int col1      = col0 + 16;
  const __bf16* wcol0 = Wt + (long)col0 * din;   // col-major weight columns
  const __bf16* wcol1 = Wt + (long)col1 * din;

  v8f c0 = {}, c1 = {};
  for (int k0 = 0; k0 < din; k0 += 32) {
    // A fragment: VGPR0-3 = K[k0 + half*16 .. +7], VGPR4-7 = K[k0 + half*16 + 8 .. +7]
    v8bf alo = *(const v8bf*)(arow + k0 + half * 16);
    v8bf ahi = *(const v8bf*)(arow + k0 + half * 16 + 8);
    v16bf a;
#pragma unroll
    for (int i = 0; i < 8; ++i) { a[i] = alo[i]; a[i + 8] = ahi[i]; }
    // B fragment: VGPR j packs K = k0 + half*16 + {2j,2j+1} -> 16 contiguous bf16
    v16bf b0 = *(const v16bf*)(wcol0 + k0 + half * 16);
    v16bf b1 = *(const v16bf*)(wcol1 + k0 + half * 16);
    c0 = __builtin_amdgcn_wmma_f32_16x16x32_bf16(false, a, false, b0, (short)0, c0,
                                                 false, false);
    c1 = __builtin_amdgcn_wmma_f32_16x16x32_bf16(false, a, false, b1, (short)0, c1,
                                                 false, false);
  }

  const float bv0 = bias[col0];
  const float bv1 = bias[col1];
  float* orow = out + (long)(mtile * 16 + half * 8) * dout;
#pragma unroll
  for (int i = 0; i < 8; ++i) {
    orow[(long)i * dout + col0] = c0[i] + bv0;
    orow[(long)i * dout + col1] = c1[i] + bv1;
  }
}

// ---------------- pooling: pooled[g] += x[node], counts[g] += 1 ----------------
__global__ void k_pool(const float* __restrict__ x, const int* __restrict__ batch,
                       float* __restrict__ pooled, float* __restrict__ counts) {
  const int C = DOUTL / 4; // 32 float4 chunks per node
  long t = (long)blockIdx.x * blockDim.x + threadIdx.x;
  if (t >= (long)NNODES * C) return;
  int node = (int)(t / C);
  int c    = (int)(t % C);
  int g = batch[node];
  const float4 v = ((const float4*)(x + (long)node * DOUTL))[c];
  float* p = pooled + (long)g * DOUTL + c * 4;
  atomAddF32(p + 0, v.x);
  atomAddF32(p + 1, v.y);
  atomAddF32(p + 2, v.z);
  atomAddF32(p + 3, v.w);
  if (c == 0) atomAddF32(counts + g, 1.0f);
}

// ---------------- head: mean, FC 128->10, log_softmax ----------------
__global__ void k_head(const float* __restrict__ pooled, const float* __restrict__ counts,
                       const float* __restrict__ Wfc, const float* __restrict__ bfc,
                       float* __restrict__ out) {
  int g = threadIdx.x;
  if (g >= NGRAPHS) return;
  float inv = 1.0f / fmaxf(counts[g], 1.0f);
  float l[OUTDIM];
#pragma unroll
  for (int o = 0; o < OUTDIM; ++o) l[o] = bfc[o];
  for (int i = 0; i < DOUTL; ++i) {
    float xv = pooled[(long)g * DOUTL + i] * inv;
#pragma unroll
    for (int o = 0; o < OUTDIM; ++o) l[o] += xv * Wfc[i * OUTDIM + o];
  }
  float m = l[0];
#pragma unroll
  for (int o = 1; o < OUTDIM; ++o) m = fmaxf(m, l[o]);
  float s = 0.0f;
#pragma unroll
  for (int o = 0; o < OUTDIM; ++o) s += expf(l[o] - m);
  float ls = logf(s);
#pragma unroll
  for (int o = 0; o < OUTDIM; ++o) out[g * OUTDIM + o] = l[o] - m - ls;
}

// =================================================================================
extern "C" void kernel_launch(void* const* d_in, const int* in_sizes, int n_in,
                              void* d_out, int out_size, void* d_ws, size_t ws_size,
                              hipStream_t stream) {
  const float* x   = (const float*)d_in[0];
  const float* W0  = (const float*)d_in[1];
  const float* b0  = (const float*)d_in[2];
  const float* W1  = (const float*)d_in[3];
  const float* b1  = (const float*)d_in[4];
  const float* W2  = (const float*)d_in[5];
  const float* b2  = (const float*)d_in[6];
  const float* W3  = (const float*)d_in[7];
  const float* b3  = (const float*)d_in[8];
  const float* Wfc = (const float*)d_in[9];
  const float* bfc = (const float*)d_in[10];
  const int* edge  = (const int*)d_in[11];
  const int* batch = (const int*)d_in[12];
  const int* src = edge;
  const int* dst = edge + NEDGES;

  // workspace carve-up (256B aligned)
  char*  ws = (char*)d_ws;
  size_t off = 0;
  auto carve = [&](size_t bytes) -> void* {
    void* p = ws + off;
    off = (off + bytes + 255) & ~(size_t)255;
    return p;
  };
  float*  xcur   = (float*)carve((size_t)NNODES * DH * 4);
  float*  agg    = (float*)carve((size_t)NNODES * DH * 4);
  __bf16* habf   = (__bf16*)carve((size_t)NNODES * DH * 2);
  __bf16* Wt0    = (__bf16*)carve((size_t)DIN0 * DH * 2);
  __bf16* Wt1    = (__bf16*)carve((size_t)DH * DH * 2);
  __bf16* Wt2    = (__bf16*)carve((size_t)DH * DH * 2);
  __bf16* Wt3    = (__bf16*)carve((size_t)DH * DOUTL * 2);
  float*  pooled = (float*)carve((size_t)NGRAPHS * DOUTL * 4);
  float*  counts = (float*)carve((size_t)NGRAPHS * 4);

  const int T = 256;

  // weight prep (bf16, transposed to column-major)
  k_wtcvt<<<(DIN0 * DH + T - 1) / T, T, 0, stream>>>(W0, Wt0, DIN0, DH);
  k_wtcvt<<<(DH * DH + T - 1) / T, T, 0, stream>>>(W1, Wt1, DH, DH);
  k_wtcvt<<<(DH * DH + T - 1) / T, T, 0, stream>>>(W2, Wt2, DH, DH);
  k_wtcvt<<<(DH * DOUTL + T - 1) / T, T, 0, stream>>>(W3, Wt3, DH, DOUTL);

  auto layer = [&](const float* xin, int din, int dout, const __bf16* Wt,
                   const float* bias, float* xout) {
    long n4 = (long)NNODES * din / 4;
    k_zero_f4<<<(unsigned)((n4 + T - 1) / T), T, 0, stream>>>((float4*)agg, n4);
    long st = (long)NEDGES * (din / 4);
    if (din == DIN0)
      k_scatter<DIN0 / 4><<<(unsigned)((st + T - 1) / T), T, 0, stream>>>(xin, src, dst, agg);
    else
      k_scatter<DH / 4><<<(unsigned)((st + T - 1) / T), T, 0, stream>>>(xin, src, dst, agg);
    k_addcvt<<<(unsigned)((n4 + T - 1) / T), T, 0, stream>>>(
        (const float4*)xin, (const float4*)agg, (v4bf*)habf, n4);
    dim3 g((NNODES / 16 + 7) / 8, dout / 32);
    k_gemm<<<g, 256, 0, stream>>>(habf, Wt, bias, xout, din, dout);
  };

  layer(x,    DIN0, DH,    Wt0, b0, xcur);
  layer(xcur, DH,   DH,    Wt1, b1, xcur);   // gemm only reads habf -> in-place OK
  layer(xcur, DH,   DH,    Wt2, b2, xcur);
  layer(xcur, DH,   DOUTL, Wt3, b3, xcur);

  // global mean pool + head
  k_zero_f4<<<((NGRAPHS * DOUTL / 4) + T - 1) / T, T, 0, stream>>>(
      (float4*)pooled, NGRAPHS * DOUTL / 4);
  k_zero_f4<<<1, NGRAPHS / 4, 0, stream>>>((float4*)counts, NGRAPHS / 4);
  long pt = (long)NNODES * (DOUTL / 4);
  k_pool<<<(unsigned)((pt + T - 1) / T), T, 0, stream>>>(xcur, batch, pooled, counts);
  k_head<<<1, 64, 0, stream>>>(pooled, counts, Wfc, bfc, (float*)d_out);
}
